// NT_Xent_22076131901579
// MI455X (gfx1250) — matrix operand order repaired and checked by reading, not compiled, not run on hardware
//
#include <hip/hip_runtime.h>
#include <hip/hip_bf16.h>
#include <math.h>

// NT-Xent (SimCLR) loss, B=4096, D=256, T=0.5 (1/T = 2).
// f16 WMMA fused GEMM->exp->rowsum; the 8192x8192 sim matrix is never
// materialized. z (f16, 4MB) is L2-resident; A tiles live in registers and
// B tiles are register double-buffered (ping-pong) to hide L2 latency.

typedef _Float16 half8  __attribute__((ext_vector_type(8)));
typedef _Float16 v16h   __attribute__((ext_vector_type(16)));
typedef float    v8f    __attribute__((ext_vector_type(8)));
typedef float    f4     __attribute__((ext_vector_type(4)));

#define B_ROWS 4096
#define D_DIM  256
#define N2     8192      // 2B rows of z
#define NT     (N2 / 16) // 512 column tiles
#define INV_T  2.0f

// ---------------------------------------------------------------------------
// Kernel A: L2-normalize rows of out0/out1, compute pos = z0.z1 (f32), and
// store z in f16 using a swizzled WMMA B-fragment layout with contiguous k per lane half:
//   tile t = (rowtile jt)*8 + kblock kb  holds 32 lanes x 16 halves;
//   lane l = khi*16 + n  <->  column (jt*16+n), K = kb*32 + khi*16 + 0..15.
// One wave handles one row pair (i, B+i). Lane L owns k = 8L..8L+7 -> exactly
// one half8 sub-chunk of one swizzled slot: kb=L>>2, khi=(L>>1)&1, half=L&1.
// ---------------------------------------------------------------------------
__global__ __launch_bounds__(256) void nt_normalize_pack(
    const float* __restrict__ out0, const float* __restrict__ out1,
    _Float16* __restrict__ zB, float* __restrict__ posdot)
{
    const int lane = threadIdx.x & 31;
    const int i    = (blockIdx.x * blockDim.x + threadIdx.x) >> 5;   // 0..4095

    const float* p0 = out0 + (size_t)i * D_DIM + lane * 8;
    const float* p1 = out1 + (size_t)i * D_DIM + lane * 8;
    f4 a0 = *(const f4*)(p0);
    f4 a1 = *(const f4*)(p0 + 4);
    f4 b0 = *(const f4*)(p1);
    f4 b1 = *(const f4*)(p1 + 4);

    float s0 = 0.f, s1 = 0.f;
#pragma unroll
    for (int e = 0; e < 4; ++e) {
        s0 += a0[e] * a0[e] + a1[e] * a1[e];
        s1 += b0[e] * b0[e] + b1[e] * b1[e];
    }
#pragma unroll
    for (int m = 16; m >= 1; m >>= 1) {
        s0 += __shfl_xor(s0, m, 32);
        s1 += __shfl_xor(s1, m, 32);
    }
    const float r0 = 1.0f / fmaxf(sqrtf(s0), 1e-12f);
    const float r1 = 1.0f / fmaxf(sqrtf(s1), 1e-12f);

    float pd = 0.f;
    half8 h0, h1;
#pragma unroll
    for (int e = 0; e < 4; ++e) {
        float z0a = a0[e] * r0, z0b = a1[e] * r0;
        float z1a = b0[e] * r1, z1b = b1[e] * r1;
        pd += z0a * z1a + z0b * z1b;
        h0[e] = (_Float16)z0a;  h0[e + 4] = (_Float16)z0b;
        h1[e] = (_Float16)z1a;  h1[e + 4] = (_Float16)z1b;
    }
#pragma unroll
    for (int m = 16; m >= 1; m >>= 1) pd += __shfl_xor(pd, m, 32);
    if (lane == 0) posdot[i] = pd;   // exact f32 positive-pair dot

    const int n       = i & 15;
    const int jt0     = i >> 4;
    const int kb      = lane >> 2;
    const int khi     = (lane >> 1) & 1;
    const int halfsel = lane & 1;
    half8* zb8 = (half8*)zB;
    zb8[((((jt0 * 8 + kb) * 32) + khi * 16 + n) << 1) + halfsel] = h0;
    const int jt1 = (B_ROWS + i) >> 4;
    zb8[((((jt1 * 8 + kb) * 32) + khi * 16 + n) << 1) + halfsel] = h1;
}

// ---------------------------------------------------------------------------
// Kernel B: fused sim row-sum. Each wave owns a 16-row strip; A fragments for
// all of K=256 (8 x v16h = 64 VGPRs) stay resident. B fragments are register
// double-buffered: tile jt+1's 16 b128 loads are issued before tile jt's
// WMMA chain, so s_wait_loadcnt has a whole tile of compute as slack.
// ---------------------------------------------------------------------------
__global__ __launch_bounds__(256) void nt_simrowsum(
    const _Float16* __restrict__ zB, float* __restrict__ denom)
{
    const int lane = threadIdx.x & 31;
    const int wave = threadIdx.x >> 5;
    const int I0   = (blockIdx.x * 8 + wave) * 16;   // strip base row
    const int m    = lane & 15;
    const int hi   = lane >> 4;

    const half8* zb8  = (const half8*)zB;
    const v16h*  zb16 = (const v16h*)zB;

    // Resident A fragments for rows I0..I0+15, K=0..255 (contiguous k per lane half, same order as B)
    v16h A[8];
    const int jtA = I0 >> 4;
#pragma unroll
    for (int kb = 0; kb < 8; ++kb) {
        const int t = jtA * 8 + kb;
        half8 lo = zb8[((t * 32 + hi * 16 + m) << 1)];          // K = kb*32 + hi*16 ..
        half8 hh = zb8[((t * 32 + hi * 16 + m) << 1) + 1];     // K = kb*32+hi*16+8 ..
        A[kb] = __builtin_shufflevector(lo, hh, 0, 1, 2, 3, 4, 5, 6, 7,
                                                8, 9, 10, 11, 12, 13, 14, 15);
    }

    float acc[8];
#pragma unroll
    for (int r = 0; r < 8; ++r) acc[r] = 0.f;

    const int rowbase = I0 + hi * 8;   // C/D: lanes 0-15 -> M=r, 16-31 -> M=8+r

    v16h B0[8], B1[8];
    {
        const v16h* bp = zb16 + lane;                      // tile 0
#pragma unroll
        for (int kb = 0; kb < 8; ++kb) B0[kb] = bp[kb * 32];
    }

    for (int jt = 0; jt < NT; jt += 2) {
        // ---- issue loads for tile jt+1 (always valid: NT is even) ----
        {
            const v16h* bp = zb16 + (size_t)(jt + 1) * 256 + lane;
#pragma unroll
            for (int kb = 0; kb < 8; ++kb) B1[kb] = bp[kb * 32];
            __builtin_prefetch((const void*)(bp + 256), 0, 3);
        }
        // ---- compute tile jt from B0 ----
        {
            v8f c = {};
#pragma unroll
            for (int kb = 0; kb < 8; ++kb)
                c = __builtin_amdgcn_wmma_f32_16x16x32_f16(
                        false, A[kb], false, B0[kb], (short)0, c, false, false);
            const int col = jt * 16 + m;
#pragma unroll
            for (int r = 0; r < 8; ++r) {
                float e = __expf(INV_T * c[r]);
                acc[r] += (rowbase + r != col) ? e : 0.f;  // drop diagonal
            }
        }
        // ---- issue loads for tile jt+2 ----
        if (jt + 2 < NT) {
            const v16h* bp = zb16 + (size_t)(jt + 2) * 256 + lane;
#pragma unroll
            for (int kb = 0; kb < 8; ++kb) B0[kb] = bp[kb * 32];
            __builtin_prefetch((const void*)(bp + 256), 0, 3);
        }
        // ---- compute tile jt+1 from B1 ----
        {
            v8f c = {};
#pragma unroll
            for (int kb = 0; kb < 8; ++kb)
                c = __builtin_amdgcn_wmma_f32_16x16x32_f16(
                        false, A[kb], false, B1[kb], (short)0, c, false, false);
            const int col = (jt + 1) * 16 + m;
#pragma unroll
            for (int r = 0; r < 8; ++r) {
                float e = __expf(INV_T * c[r]);
                acc[r] += (rowbase + r != col) ? e : 0.f;
            }
        }
    }

    // Row sums: reduce across the 16 lanes of each half (masks < 16)
#pragma unroll
    for (int r = 0; r < 8; ++r) {
        float v = acc[r];
        v += __shfl_xor(v, 1, 32);
        v += __shfl_xor(v, 2, 32);
        v += __shfl_xor(v, 4, 32);
        v += __shfl_xor(v, 8, 32);
        acc[r] = v;
    }
    if (m == 0) {   // lane 0 -> rows I0..I0+7, lane 16 -> rows I0+8..I0+15
#pragma unroll
        for (int r = 0; r < 8; ++r) denom[rowbase + r] = acc[r];
    }
}

// ---------------------------------------------------------------------------
// Kernel C: loss = mean_i( log(denom_i) - pos_{i mod B} / T )
// ---------------------------------------------------------------------------
__global__ __launch_bounds__(256) void nt_loss(
    const float* __restrict__ denom, const float* __restrict__ posdot,
    float* __restrict__ out)
{
    __shared__ float red[8];
    float s = 0.f;
    for (int i = threadIdx.x; i < N2; i += 256)
        s += logf(denom[i]) - INV_T * posdot[i & (B_ROWS - 1)];
#pragma unroll
    for (int m = 16; m >= 1; m >>= 1) s += __shfl_xor(s, m, 32);
    if ((threadIdx.x & 31) == 0) red[threadIdx.x >> 5] = s;
    __syncthreads();
    if (threadIdx.x == 0) {
        float t = 0.f;
#pragma unroll
        for (int w = 0; w < 8; ++w) t += red[w];
        out[0] = t * (1.0f / (float)N2);
    }
}

extern "C" void kernel_launch(void* const* d_in, const int* in_sizes, int n_in,
                              void* d_out, int out_size, void* d_ws, size_t ws_size,
                              hipStream_t stream)
{
    const float* out0 = (const float*)d_in[0];
    const float* out1 = (const float*)d_in[1];
    float* loss = (float*)d_out;

    char* ws = (char*)d_ws;
    _Float16* zB  = (_Float16*)ws;                               // 4 MiB
    float* posdot = (float*)(ws + (size_t)N2 * D_DIM * 2);       // 16 KiB
    float* denom  = posdot + B_ROWS;                             // 32 KiB

    nt_normalize_pack<<<B_ROWS / 8, 256, 0, stream>>>(out0, out1, zB, posdot);
    nt_simrowsum<<<N2 / 128, 256, 0, stream>>>(zB, denom);
    nt_loss<<<1, 256, 0, stream>>>(denom, posdot, loss);
}
